// LSTMGenerator_47476568490595
// MI455X (gfx1250) — matrix operand order repaired and checked by reading, not compiled, not run on hardware
//
#include <hip/hip_runtime.h>
#include <hip/hip_bf16.h>

// ---------------------------------------------------------------------------
// 3-layer LSTM (H = 32, 64, 128) + linear head for MI455X (gfx1250).
// Batch-parallel persistent scan: 16 workgroups x 16 batch rows (one WMMA
// M-tile), weights + h/c state resident in LDS, recurrent + input GEMMs fused
// with v_wmma_f32_16x16x32_bf16 (f32 accumulate). Layers 1/2 stream their
// bf16 input tiles via the Tensor Data Mover (tensor_load_to_lds), double-
// buffered and pipelined one step ahead, synchronized with s_wait_tensorcnt.
// ---------------------------------------------------------------------------

typedef __attribute__((ext_vector_type(16))) __bf16 bf16x16;
typedef __attribute__((ext_vector_type(8)))  __bf16 bf16x8;
typedef __attribute__((ext_vector_type(8)))  float  f32x8;
typedef unsigned int uint32x4 __attribute__((ext_vector_type(4)));
typedef int          int32x4  __attribute__((ext_vector_type(4)));
typedef int          int32x8  __attribute__((ext_vector_type(8)));

#define B_SZ 256
#define T_SZ 1024

#if __has_builtin(__builtin_amdgcn_tensor_load_to_lds)
#define HAVE_TDM 1
#else
#define HAVE_TDM 0
#endif

__device__ __forceinline__ float sigmoid_fast(float x) {
  return 1.0f / (1.0f + __expf(-x));
}
__device__ __forceinline__ float tanh_fast(float x) {
  x = fminf(fmaxf(x, -15.0f), 15.0f);
  float e = __expf(2.0f * x);
  return (e - 1.0f) / (e + 1.0f);
}

__device__ __forceinline__ void wait_tensorcnt0() {
#if __has_builtin(__builtin_amdgcn_s_wait_tensorcnt)
  __builtin_amdgcn_s_wait_tensorcnt(0);
#else
  asm volatile("s_wait_tensorcnt 0x0" ::: "memory");
#endif
}

#if HAVE_TDM
// Issue a 2D TDM load: `rows` rows of `row_elems` bf16 elements, source rows
// `row_stride_elems` apart, into LDS at byte offset `lds_off`.
// D# packing per CDNA5 ISA 8.3/8.4 (group0: flags/lds/global addr; group1:
// data_size=2B, tensor dims, tile dims, strides). Groups 2/3 zero (2D tile).
__device__ __forceinline__ void tdm_load_2d_bf16(
    unsigned long long gaddr, unsigned int lds_off, unsigned int rows,
    unsigned int row_elems, unsigned long long row_stride_elems) {
  uint32x4 g0;
  g0[0] = 1u;                                               // count=1 (valid), user mode
  g0[1] = lds_off;                                          // lds_addr (bytes)
  g0[2] = (unsigned int)(gaddr & 0xFFFFFFFFull);            // global_addr[31:0]
  g0[3] = (unsigned int)((gaddr >> 32) & 0x1FFFFFFull)      // global_addr[56:32]
        | (2u << 30);                                       // type=2 ("image")
  int32x8 g1;
  g1[0] = (int)(1u << 16);                                  // data_size=1 (2 bytes)
  g1[1] = (int)((row_elems & 0xFFFFu) << 16);               // tensor_dim0[15:0]
  g1[2] = (int)(((row_elems >> 16) & 0xFFFFu)               // tensor_dim0[31:16]
        | ((rows & 0xFFFFu) << 16));                        // tensor_dim1[15:0]
  g1[3] = (int)((row_elems & 0xFFFFu) << 16);               // tile_dim0
  g1[4] = (int)(rows & 0xFFFFu);                            // tile_dim1 (tile_dim2=0)
  g1[5] = (int)(row_stride_elems & 0xFFFFFFFFull);          // tensor_dim0_stride[31:0]
  g1[6] = (int)((row_stride_elems >> 32) & 0xFFFFull);      // stride[47:32]
  g1[7] = 0;
  int32x4 z4 = {0, 0, 0, 0};
#if defined(__clang_major__) && __clang_major__ >= 23
  int32x8 z8 = {0, 0, 0, 0, 0, 0, 0, 0};
  __builtin_amdgcn_tensor_load_to_lds(g0, g1, z4, z4, z8, 0);
#else
  __builtin_amdgcn_tensor_load_to_lds(g0, g1, z4, z4, 0);
#endif
}
#endif  // HAVE_TDM

// A fragment (16x32 bf16, MxK) from row-major LDS [16][C]:
// lane l: M = l&15 ; 16 contiguous K at K = kt*32 + (l>=16 ? 16:0).
__device__ __forceinline__ bf16x16 load_frag_A(const __bf16* base, int C, int kt) {
  int lane = threadIdx.x & 31;
  int m  = lane & 15;
  int c0 = kt * 32 + ((lane >> 4) << 4);
  const __bf16* p = base + m * C + c0;
  bf16x8 lo = *(const bf16x8*)(p);
  bf16x8 hi = *(const bf16x8*)(p + 8);
  return __builtin_shufflevector(lo, hi, 0,1,2,3,4,5,6,7,8,9,10,11,12,13,14,15);
}

// B fragment (32x16 bf16, KxN), B[K][N] = W[N][K]; W row-major [4H][C].
// lane l: N = l&15 ; 16 contiguous K at K = kt*32 + (l>=16 ? 16:0).
__device__ __forceinline__ bf16x16 load_frag_B(const __bf16* W, int C, int nt, int kt) {
  int lane = threadIdx.x & 31;
  const __bf16* p = W + (nt * 16 + (lane & 15)) * C + kt * 32 + (lane >> 4) * 16;
  bf16x8 lo = *(const bf16x8*)(p);
  bf16x8 hi = *(const bf16x8*)(p + 8);
  return __builtin_shufflevector(lo, hi, 0,1,2,3,4,5,6,7,8,9,10,11,12,13,14,15);
}

// LDS budget per layer (bytes); nxbuf = 1 (manual) or 2 (TDM double buffer)
constexpr size_t lds_bytes(int H, int IPAD, int nxbuf) {
  size_t G4 = 4 * (size_t)H;
  return G4 * IPAD * 2                   // Wih bf16 (padded)
       + G4 * H * 2                      // Whh bf16
       + (size_t)nxbuf * 16 * IPAD * 2   // x tile(s) bf16
       + 16 * (size_t)H * 2              // h state bf16
       + 16 * (size_t)H * 4              // c state f32
       + 16 * G4 * 4                     // g scratch f32
       + G4 * 4;                         // bias f32
}

template <int H, int I, int IPAD, bool IN_F32, bool WANT_TDM>
__global__ __launch_bounds__(256)
void lstm_scan_kernel(const void*  __restrict__ xin,    // [B,T,I] f32 or bf16
                      const float* __restrict__ Wih,    // [4H,I]
                      const float* __restrict__ Whh,    // [4H,H]
                      const float* __restrict__ bias,   // [4H]
                      const float* __restrict__ h0,     // [B,H] or null
                      const float* __restrict__ c0,     // [B,H] or null
                      __bf16*      __restrict__ out_bf, // [B,T,H] or null
                      float*       __restrict__ out_f32)// [B,T,H] or null
{
  constexpr int  G4     = 4 * H;
  constexpr int  KX     = IPAD / 32;     // K-tiles, input proj
  constexpr int  KH     = H / 32;        // K-tiles, recurrent proj
  constexpr int  NTILES = G4 / 16;       // N-tiles across 4H gates
  constexpr int  NW     = 8;             // waves per block (wave32)
  constexpr int  NTW    = NTILES / NW;   // N-tiles per wave
  constexpr bool USE_TDM = WANT_TDM && !IN_F32 && (I == IPAD) && (HAVE_TDM != 0);
  constexpr int  NXBUF  = USE_TDM ? 2 : 1;
  static_assert(IPAD % 32 == 0 && H % 32 == 0 && NTILES % NW == 0, "shape");

  extern __shared__ char smem_raw[];
  __bf16* Wih_s  = (__bf16*)smem_raw;               // [4H][IPAD]
  __bf16* Whh_s  = Wih_s + G4 * IPAD;               // [4H][H]
  __bf16* x_s    = Whh_s + (size_t)G4 * H;          // [NXBUF][16][IPAD]
  __bf16* h_s    = x_s + NXBUF * 16 * IPAD;         // [16][H]
  float*  c_s    = (float*)(h_s + 16 * H);          // [16][H]
  float*  g_s    = c_s + 16 * H;                    // [16][4H]
  float*  bias_s = g_s + 16 * G4;                   // [4H]
  // byte offset of x_s within the (dynamic-only) LDS allocation
  constexpr unsigned XOFF = (unsigned)((G4 * IPAD + G4 * H) * 2);

  const int tid  = threadIdx.x;
  const int wave = tid >> 5;
  const int m0   = blockIdx.x * 16;                 // first batch row

  // ---- stage weights into LDS as bf16 (zero-pad Wih cols I..IPAD) ----
  for (int idx = tid; idx < G4 * IPAD; idx += 256) {
    int r = idx / IPAD, c = idx % IPAD;
    Wih_s[idx] = (__bf16)((c < I) ? Wih[r * I + c] : 0.0f);
  }
  for (int idx = tid; idx < G4 * H; idx += 256) Whh_s[idx] = (__bf16)Whh[idx];
  for (int idx = tid; idx < G4; idx += 256) bias_s[idx] = bias[idx];

  // ---- init h, c state; zero the constant x padding region once ----
  for (int idx = tid; idx < 16 * H; idx += 256) {
    int m = idx / H, j = idx % H;
    h_s[idx] = (__bf16)(h0 ? h0[(m0 + m) * H + j] : 0.0f);
    c_s[idx] = c0 ? c0[(m0 + m) * H + j] : 0.0f;
  }
  if (!USE_TDM && I < IPAD) {
    for (int idx = tid; idx < 16 * (IPAD - I); idx += 256) {
      int m = idx / (IPAD - I), c = I + idx % (IPAD - I);
      x_s[m * IPAD + c] = (__bf16)0.0f;
    }
  }

#if HAVE_TDM
  if constexpr (USE_TDM) {
    if (wave == 0) {  // pipeline prologue: t = 0 tile into buffer 0
      unsigned long long ga = (unsigned long long)(uintptr_t)xin +
                              ((unsigned long long)m0 * T_SZ) * I * 2ull;
      tdm_load_2d_bf16(ga, XOFF, 16u, (unsigned)I,
                       (unsigned long long)T_SZ * I);
    }
  }
#endif
  __syncthreads();

  for (int t = 0; t < T_SZ; ++t) {
    // ---- stage 1: x[:,t,:] tile -> bf16 LDS ----
    if constexpr (!USE_TDM) {
      const float*  xf = (const float*)xin;
      const __bf16* xh = (const __bf16*)xin;
      for (int idx = tid; idx < 16 * I; idx += 256) {
        int m = idx / I, c = idx % I;
        size_t off = ((size_t)(m0 + m) * T_SZ + t) * I + c;
        float v = IN_F32 ? xf[off] : (float)xh[off];
        x_s[m * IPAD + c] = (__bf16)v;
      }
      if (t + 1 < T_SZ) {  // prefetch next step, one touch per 128B chunk
        constexpr int CH = (I * (IN_F32 ? 4 : 2) + 127) / 128;
        for (int idx = tid; idx < 16 * CH; idx += 256) {
          int m = idx / CH, ch = idx % CH;
          size_t off = ((size_t)(m0 + m) * T_SZ + (t + 1)) * I;
          const char* p = IN_F32 ? (const char*)(xf + off) : (const char*)(xh + off);
          __builtin_prefetch(p + ch * 128, 0, 1);
        }
      }
    } else {
      if (wave == 0) wait_tensorcnt0();  // tile for step t landed in LDS
    }
    __syncthreads();  // barrier A: x tile ready, h_s(t-1) visible

    // ---- stage 2: g = x_t * Wih^T + h_{t-1} * Whh^T  (WMMA, f32 accum) ----
    const __bf16* xb = x_s + (USE_TDM ? (t & 1) * (16 * IPAD) : 0);
    bf16x16 a_x[KX];
    bf16x16 a_h[KH];
#pragma unroll
    for (int k = 0; k < KX; ++k) a_x[k] = load_frag_A(xb, IPAD, k);
#pragma unroll
    for (int k = 0; k < KH; ++k) a_h[k] = load_frag_A(h_s, H, k);

    const int lane = tid & 31;
    const int mrow = (lane >> 4) * 8;
#pragma unroll
    for (int j = 0; j < NTW; ++j) {
      const int nt = wave * NTW + j;
      f32x8 acc = {};
#pragma unroll
      for (int k = 0; k < KX; ++k) {
        bf16x16 bfr = load_frag_B(Wih_s, IPAD, nt, k);
        acc = __builtin_amdgcn_wmma_f32_16x16x32_bf16(
            false, a_x[k], false, bfr, (short)0, acc, false, false);
      }
#pragma unroll
      for (int k = 0; k < KH; ++k) {
        bf16x16 bfr = load_frag_B(Whh_s, H, nt, k);
        acc = __builtin_amdgcn_wmma_f32_16x16x32_bf16(
            false, a_h[k], false, bfr, (short)0, acc, false, false);
      }
      // C/D layout: VGPR r -> row M = r + 8*(lane>=16), col = nt*16 + (lane&15)
      const int col = nt * 16 + (lane & 15);
#pragma unroll
      for (int r = 0; r < 8; ++r) g_s[(mrow + r) * G4 + col] = acc[r];
    }
    __syncthreads();  // barrier B: g complete; x/h fully consumed

    // ---- stage 3: gates (i,f,g,o), state update, emit h_t ----
#if HAVE_TDM
    if constexpr (USE_TDM) {
      if (wave == 0 && t + 1 < T_SZ) {  // prefetch t+1 into the other buffer
        unsigned long long ga = (unsigned long long)(uintptr_t)xin +
            ((unsigned long long)m0 * T_SZ + (t + 1)) * I * 2ull;
        tdm_load_2d_bf16(ga, XOFF + (unsigned)(((t + 1) & 1) * 16 * IPAD * 2),
                         16u, (unsigned)I, (unsigned long long)T_SZ * I);
      }
    }
#endif
    for (int idx = tid; idx < 16 * H; idx += 256) {
      int m = idx / H, j = idx % H;
      const float* gr = g_s + m * G4;
      float iv = sigmoid_fast(gr[j]         + bias_s[j]);
      float fv = sigmoid_fast(gr[H + j]     + bias_s[H + j]);
      float gv = tanh_fast   (gr[2 * H + j] + bias_s[2 * H + j]);
      float ov = sigmoid_fast(gr[3 * H + j] + bias_s[3 * H + j]);
      float cv = fv * c_s[idx] + iv * gv;
      float hv = ov * tanh_fast(cv);
      c_s[idx] = cv;
      h_s[idx] = (__bf16)hv;
      size_t off = ((size_t)(m0 + m) * T_SZ + t) * H + j;
      if (out_bf)  out_bf[off]  = (__bf16)hv;
      if (out_f32) out_f32[off] = hv;
    }
    __syncthreads();  // barrier A of step t+1 separates h_s writes from reads
  }
}

// out[b,t] = tanh( dot(r2[b,t,:], Wl) + bl ) ;  H2 = 128, OUT = 1
__global__ __launch_bounds__(256)
void final_linear_kernel(const float* __restrict__ r2,
                         const float* __restrict__ Wl,
                         const float* __restrict__ bl,
                         float* __restrict__ out, int n) {
  int idx = blockIdx.x * blockDim.x + threadIdx.x;
  if (idx >= n) return;
  const float4* row = (const float4*)(r2 + (size_t)idx * 128);
  const float4* w   = (const float4*)Wl;
  float s = bl[0];
#pragma unroll
  for (int j = 0; j < 32; ++j) {
    float4 a = row[j], b = w[j];
    s += a.x * b.x + a.y * b.y + a.z * b.z + a.w * b.w;
  }
  out[idx] = tanh_fast(s);
}

extern "C" void kernel_launch(void* const* d_in, const int* in_sizes, int n_in,
                              void* d_out, int out_size, void* d_ws, size_t ws_size,
                              hipStream_t stream) {
  const float* x    = (const float*)d_in[0];   // [256,1024,100]
  const float* h0   = (const float*)d_in[1];   // [1,256,32]
  const float* c0   = (const float*)d_in[2];
  const float* Wih0 = (const float*)d_in[3];
  const float* Whh0 = (const float*)d_in[4];
  const float* b0   = (const float*)d_in[5];
  const float* Wih1 = (const float*)d_in[6];
  const float* Whh1 = (const float*)d_in[7];
  const float* b1   = (const float*)d_in[8];
  const float* Wih2 = (const float*)d_in[9];
  const float* Whh2 = (const float*)d_in[10];
  const float* b2   = (const float*)d_in[11];
  const float* Wl   = (const float*)d_in[12];  // [1,128]
  const float* bl   = (const float*)d_in[13];  // [1]

  // Output tuple (out, r) flattened: out [B,T,1] then r [B,T,128]
  float* out_lin = (float*)d_out;
  float* r2_out  = (float*)d_out + (size_t)B_SZ * T_SZ;

  // Workspace: bf16 inter-layer activations
  __bf16* r0 = (__bf16*)d_ws;                              // [B,T,32]
  __bf16* r1 = r0 + (size_t)B_SZ * T_SZ * 32;              // [B,T,64]

  const dim3 grid(B_SZ / 16), blk(256);
  const size_t s0 = lds_bytes(32, 128, 1);
  const size_t s1 = lds_bytes(64, 32, 2);
  const size_t s2 = lds_bytes(128, 64, 2);

  // Allow >64KB dynamic LDS (capture-safe; not stream-ordered)
  hipFuncSetAttribute((const void*)lstm_scan_kernel<32, 100, 128, true, false>,
                      hipFuncAttributeMaxDynamicSharedMemorySize, (int)s0);
  hipFuncSetAttribute((const void*)lstm_scan_kernel<64, 32, 32, false, true>,
                      hipFuncAttributeMaxDynamicSharedMemorySize, (int)s1);
  hipFuncSetAttribute((const void*)lstm_scan_kernel<128, 64, 64, false, true>,
                      hipFuncAttributeMaxDynamicSharedMemorySize, (int)s2);

  // Layer 0: fp32 input (manual convert path), h0/c0 provided, emit bf16 r0
  lstm_scan_kernel<32, 100, 128, true, false><<<grid, blk, s0, stream>>>(
      x, Wih0, Whh0, b0, h0, c0, r0, nullptr);
  // Layer 1: bf16 input via TDM, zero state, emit bf16 r1
  lstm_scan_kernel<64, 32, 32, false, true><<<grid, blk, s1, stream>>>(
      r0, Wih1, Whh1, b1, nullptr, nullptr, r1, nullptr);
  // Layer 2: bf16 input via TDM, zero state, emit fp32 r2 into d_out
  lstm_scan_kernel<128, 64, 64, false, true><<<grid, blk, s2, stream>>>(
      r1, Wih2, Whh2, b2, nullptr, nullptr, nullptr, r2_out);

  // Head: out = tanh(r2 @ Wl^T + bl)
  const int n = B_SZ * T_SZ;
  final_linear_kernel<<<(n + 255) / 256, 256, 0, stream>>>(r2_out, Wl, bl, out_lin, n);
}
